// TDEER_25331717111817
// MI455X (gfx1250) — matrix operand order repaired and checked by reading, not compiled, not run on hardware
//
#include <hip/hip_runtime.h>

#define B_  32
#define S_  1024
#define H_  768
#define BS_ (B_*S_)

typedef float  v8f   __attribute__((ext_vector_type(8)));
typedef __bf16 v16bf __attribute__((ext_vector_type(16)));
typedef unsigned short u16;
typedef u16 u16x8  __attribute__((ext_vector_type(8)));
typedef u16 u16x16 __attribute__((ext_vector_type(16)));

// f32 -> bf16 round-to-nearest-even
__device__ __forceinline__ u16 f2bf(float f) {
    unsigned u = __float_as_uint(f);
    u += 0x7fffu + ((u >> 16) & 1u);
    return (u16)(u >> 16);
}

__device__ __forceinline__ v16bf pack2(u16x8 lo, u16x8 hi) {
    u16x16 t = __builtin_shufflevector(lo, hi, 0,1,2,3,4,5,6,7,8,9,10,11,12,13,14,15);
    return __builtin_bit_cast(v16bf, t);
}

// A fragment: 16x32 bf16 tile, src row-major, leading dim ld (elements).
// Lane l holds row (l&15); K chunks at k0+8*(l>>4) and +16 (ISA 16-bit A layout).
__device__ __forceinline__ v16bf load_a(const u16* src, int ld, int k0, int lane) {
    int row = lane & 15;
    int kb  = k0 + ((lane >> 4) << 3);
    const u16* p = src + row * ld + kb;
    u16x8 lo = *(const u16x8*)p;
    u16x8 hi = *(const u16x8*)(p + 16);
    return pack2(lo, hi);
}

// B fragment: B[k,n] = src[n*ld + k] (src rows indexed by n, contiguous in k).
// Lane l holds col n0+(l&15); K chunks at k0+8*(l>>4) and +16 (same order as A).
__device__ __forceinline__ v16bf load_b(const u16* src, int ld, int n0, int k0, int lane) {
    int n  = n0 + (lane & 15);
    int kb = k0 + ((lane >> 4) << 3);
    const u16* p = src + n * ld + kb;
    u16x8 lo = *(const u16x8*)p;
    u16x8 hi = *(const u16x8*)(p + 16);
    return pack2(lo, hi);
}

__device__ __forceinline__ v8f wmma_bf16(v16bf a, v16bf b, v8f c) {
    return __builtin_amdgcn_wmma_f32_16x16x32_bf16(false, a, false, b, (short)0, c, false, false);
}

// ---------------- kernel 1: weights f32 -> bf16 ----------------
__global__ void cvt_w_kernel(const float* Wq, const float* Wk, const float* Wv, u16* dst) {
    int i = blockIdx.x * blockDim.x + threadIdx.x;
    if (i < H_ * H_) {
        dst[i]               = f2bf(Wq[i]);
        dst[H_ * H_ + i]     = f2bf(Wk[i]);
        dst[2 * H_ * H_ + i] = f2bf(Wv[i]);
    }
}

// ---------------- kernel 2: conditional-LN weight/bias per batch ----------------
__global__ void cond_kernel(const int* relation, const float* lhs,
                            const int* sub_head, const int* sub_tail,
                            const float* rel_emb, const float* W_rf, const float* b_rf,
                            const float* gamma, const float* beta,
                            const float* W_wd, const float* W_bd,
                            float* wgt, float* bia) {
    __shared__ float relf[H_], subf[H_];
    int b = blockIdx.x, tid = threadIdx.x;
    int r  = relation[b];
    int sh = sub_head[b], st = sub_tail[b];
    const float* emb = rel_emb + (long)r * H_;
    for (int h = tid; h < H_; h += blockDim.x) {
        float acc = b_rf[h];
        const float* wr = W_rf + (long)h * H_;
        for (int k = 0; k < H_; k++) acc += emb[k] * wr[k];
        relf[h] = fminf(fmaxf(acc, 0.f), 6.f);   // ReLU6
        subf[h] = 0.5f * (lhs[((long)b * S_ + sh) * H_ + h] +
                          lhs[((long)b * S_ + st) * H_ + h]);
    }
    __syncthreads();
    for (int h = tid; h < H_; h += blockDim.x) {
        float w = gamma[h], bo = beta[h];
        const float* wd = W_wd + (long)h * 2 * H_;
        const float* bd = W_bd + (long)h * 2 * H_;
        for (int j = 0; j < H_; j++) { w += relf[j] * wd[j];       bo += relf[j] * bd[j]; }
        for (int j = 0; j < H_; j++) { w += subf[j] * wd[H_ + j];  bo += subf[j] * bd[H_ + j]; }
        wgt[b * H_ + h] = w;
        bia[b * H_ + h] = bo;
    }
}

// ---------------- kernel 3: conditional layer norm ----------------
__global__ void ln_kernel(const float* lhs, const float* wgt, const float* bia,
                          float* xf, u16* xb) {
    __shared__ float s1[256], s2[256];
    long tok = blockIdx.x;
    int  b   = (int)(tok >> 10);
    int  tid = threadIdx.x;
    const float* row = lhs + tok * H_;
    float v[3], s = 0.f, q = 0.f;
#pragma unroll
    for (int i = 0; i < 3; i++) { v[i] = row[tid + i * 256]; s += v[i]; q += v[i] * v[i]; }
    s1[tid] = s; s2[tid] = q; __syncthreads();
    for (int off = 128; off > 0; off >>= 1) {
        if (tid < off) { s1[tid] += s1[tid + off]; s2[tid] += s2[tid + off]; }
        __syncthreads();
    }
    float mean = s1[0] * (1.f / H_);
    float var  = s2[0] * (1.f / H_) - mean * mean;
    float rstd = rsqrtf(var + 1e-12f);
    float* xro = xf + tok * H_;
    u16*   xbo = xb + tok * H_;
#pragma unroll
    for (int i = 0; i < 3; i++) {
        int h = tid + i * 256;
        float x = (v[i] - mean) * rstd * wgt[b * H_ + h] + bia[b * H_ + h];
        xro[h] = x;
        xbo[h] = f2bf(x);
    }
}

// ---------------- kernel 4: QKV projection (WMMA, 4-way N-blocked) ----------------
__global__ void __launch_bounds__(128) qkv_kernel(const u16* xb, const u16* Wbf,
        const float* bq, const float* bk, const float* bv,
        u16* qb, u16* kb, u16* vT) {
    int lane = threadIdx.x & 31;
    int wave = threadIdx.x >> 5;
    long m0  = (long)blockIdx.x * 16;         // row tile in [BS]
    int which = blockIdx.y;                   // 0=q 1=k 2=v
    const u16* A = xb + m0 * H_;
    const u16* W = Wbf + (long)which * H_ * H_;
    const float* bias = (which == 0) ? bq : (which == 1) ? bk : bv;
    int N  = lane & 15;
    int mh = (lane >> 4) << 3;
    for (int ng = wave; ng < H_ / 64; ng += 4) {   // 12 groups of 4 n-tiles
        int n0 = ng * 64;
        v8f acc[4];
#pragma unroll
        for (int j = 0; j < 4; j++) acc[j] = (v8f){0.f,0.f,0.f,0.f,0.f,0.f,0.f,0.f};
        for (int k0 = 0; k0 < H_; k0 += 32) {
            v16bf a = load_a(A, H_, k0, lane);     // one A, four B: 4x reuse
#pragma unroll
            for (int j = 0; j < 4; j++) {
                v16bf bm = load_b(W, H_, n0 + 16 * j, k0, lane);
                acc[j] = wmma_bf16(a, bm, acc[j]);
            }
        }
#pragma unroll
        for (int j = 0; j < 4; j++) {
            int n = n0 + 16 * j + N;
            float bn = bias[n];
            if (which < 2) {
                u16* out = (which == 0) ? qb : kb;        // row-major [m,h]
#pragma unroll
                for (int r = 0; r < 8; r++) {
                    float f = acc[j][r] + bn; f = f > 0.f ? f : 0.f;
                    out[(m0 + mh + r) * H_ + n] = f2bf(f);
                }
            } else {                                      // V transposed: [b][h][t]
                int bidx = (int)(m0 >> 10);
                int s0   = (int)(m0 & 1023) + mh;
                u16x8 pk;
#pragma unroll
                for (int r = 0; r < 8; r++) {
                    float f = acc[j][r] + bn; f = f > 0.f ? f : 0.f;
                    pk[r] = f2bf(f);
                }
                *(u16x8*)(vT + ((long)bidx * H_ + n) * S_ + s0) = pk;
            }
        }
    }
}

// ---------------- kernel 5: attention + residual + output projection ----------------
__global__ void __launch_bounds__(128) attn_kernel(const u16* qb, const u16* kb, const u16* vT,
        const float* xf, const float* mask, const float* W_obj, const float* b_obj,
        float* pred, float* hid) {
    extern __shared__ float eLDS[];               // 4 waves * 16 rows * S floats = 256 KB
    int lane = threadIdx.x & 31;
    int wave = threadIdx.x >> 5;
    int bidx = blockIdx.y;
    int s0   = (blockIdx.x * 4 + wave) * 16;
    long gm0 = (long)bidx * S_ + s0;
    float* E = eLDS + wave * 16 * S_;             // this wave's private 16xS f32 tile
    const u16* Q = qb + gm0 * H_;
    const u16* K = kb + (long)bidx * S_ * H_;
    int N  = lane & 15;
    int mh = (lane >> 4) << 3;

    // ---- pass 1: e = q.kT + mask bias (f32 in LDS), 4 t-tiles per A fragment
    float rmax[8];
#pragma unroll
    for (int r = 0; r < 8; r++) rmax[r] = -3.0e38f;
    for (int t0 = 0; t0 < S_; t0 += 64) {
        v8f acc[4];
#pragma unroll
        for (int j = 0; j < 4; j++) acc[j] = (v8f){0.f,0.f,0.f,0.f,0.f,0.f,0.f,0.f};
        for (int k0 = 0; k0 < H_; k0 += 32) {
            v16bf a = load_a(Q, H_, k0, lane);
#pragma unroll
            for (int j = 0; j < 4; j++) {
                v16bf bm = load_b(K, H_, t0 + 16 * j, k0, lane);   // B[k,t] = k[t,k]
                acc[j] = wmma_bf16(a, bm, acc[j]);
            }
        }
#pragma unroll
        for (int j = 0; j < 4; j++) {
            int t = t0 + 16 * j + N;
            float mb = -1.0e10f * (1.f - mask[bidx * S_ + t]);
#pragma unroll
            for (int r = 0; r < 8; r++) {
                float e = acc[j][r] + mb;
                E[(mh + r) * S_ + t] = e;
                rmax[r] = fmaxf(rmax[r], e);
            }
        }
    }
#pragma unroll
    for (int r = 0; r < 8; r++)
#pragma unroll
        for (int m = 1; m < 16; m <<= 1) rmax[r] = fmaxf(rmax[r], __shfl_xor(rmax[r], m, 32));

    // ---- pass 2: exponentiate and re-pack P to bf16 IN PLACE (row-sequential).
    // DS ops from one wave are in-order; bf16 write for (row,t) lands at byte
    // 2*(row*S+t) and only clobbers f32 indices already consumed, so the f32->bf16
    // in-place repack is safe when rows are processed 0..15 and t ascending.
    u16* Pb = (u16*)E;                            // bf16 view, same LDS block
    float rsum[8] = {0.f,0.f,0.f,0.f,0.f,0.f,0.f,0.f};
#pragma unroll
    for (int rr = 0; rr < 16; rr++) {
        float rm = __shfl(rmax[rr & 7], (rr >> 3) << 4, 32);   // row max broadcast
        const float* Er = E + rr * S_;
        u16* Pr = Pb + rr * S_;
        float s = 0.f;
        for (int t0 = 0; t0 < S_; t0 += 32) {
            float p = __expf(Er[t0 + lane] - rm);
            s += p;
            Pr[t0 + lane] = f2bf(p);
        }
#pragma unroll
        for (int m = 1; m < 32; m <<= 1) s += __shfl_xor(s, m, 32);
        if ((rr >> 3) == (lane >> 4)) rsum[rr & 7] = s;        // keep my half's rows
    }
    float rinv[8];
#pragma unroll
    for (int r = 0; r < 8; r++) rinv[r] = 1.f / rsum[r];
    __syncthreads();

    // ---- pass 3: hidden = (P/rowsum) @ V + x ; fused pred = hidden @ W_obj + b_obj
    // A fragments are now plain bf16 ds_load_b128 pairs: no VALU converts in hot loop.
    const u16*  V = vT + (long)bidx * H_ * S_;
    const float* X = xf + gm0 * H_;
    float pp[8] = {0.f,0.f,0.f,0.f,0.f,0.f,0.f,0.f};
    for (int n0 = 0; n0 < H_; n0 += 64) {
        v8f acc[4];
#pragma unroll
        for (int j = 0; j < 4; j++) acc[j] = (v8f){0.f,0.f,0.f,0.f,0.f,0.f,0.f,0.f};
        for (int t0 = 0; t0 < S_; t0 += 32) {
            v16bf a = load_a(Pb, S_, t0, lane);   // P bf16 from LDS
#pragma unroll
            for (int j = 0; j < 4; j++) {
                v16bf bm = load_b(V, S_, n0 + 16 * j, t0, lane);   // B[t,h] = vT[h,t]
                acc[j] = wmma_bf16(a, bm, acc[j]);
            }
        }
#pragma unroll
        for (int j = 0; j < 4; j++) {
            int n = n0 + 16 * j + N;
            float wo = W_obj[n];
#pragma unroll
            for (int r = 0; r < 8; r++) {
                float h = acc[j][r] * rinv[r] + X[(mh + r) * H_ + n];
                hid[(gm0 + mh + r) * H_ + n] = h;
                pp[r] += h * wo;
            }
        }
    }
#pragma unroll
    for (int r = 0; r < 8; r++)
#pragma unroll
        for (int m = 1; m < 16; m <<= 1) pp[r] += __shfl_xor(pp[r], m, 32);
    if (N == 0) {
        float bo = b_obj[0];
#pragma unroll
        for (int r = 0; r < 8; r++) pred[gm0 + mh + r] = pp[r] + bo;
    }
}

extern "C" void kernel_launch(void* const* d_in, const int* in_sizes, int n_in,
                              void* d_out, int out_size, void* d_ws, size_t ws_size,
                              hipStream_t stream) {
    (void)in_sizes; (void)n_in; (void)out_size; (void)ws_size;
    const int*   relation = (const int*)  d_in[0];
    const float* lhs      = (const float*)d_in[1];
    const int*   sub_head = (const int*)  d_in[2];
    const int*   sub_tail = (const int*)  d_in[3];
    const float* mask     = (const float*)d_in[4];
    const float* rel_emb  = (const float*)d_in[5];
    const float* W_rf     = (const float*)d_in[6];
    const float* b_rf     = (const float*)d_in[7];
    const float* gamma    = (const float*)d_in[8];
    const float* beta     = (const float*)d_in[9];
    const float* W_wd     = (const float*)d_in[10];
    const float* W_bd     = (const float*)d_in[11];
    const float* Wq       = (const float*)d_in[12];
    const float* bq       = (const float*)d_in[13];
    const float* Wk       = (const float*)d_in[14];
    const float* bk       = (const float*)d_in[15];
    const float* Wv       = (const float*)d_in[16];
    const float* bv       = (const float*)d_in[17];
    const float* W_obj    = (const float*)d_in[18];
    const float* b_obj    = (const float*)d_in[19];

    char* ws = (char*)d_ws;
    size_t off = 0;
    auto alloc = [&](size_t bytes) { void* p = ws + off; off += (bytes + 255) & ~(size_t)255; return p; };
    float* wgt = (float*)alloc((size_t)B_ * H_ * 4);
    float* bia = (float*)alloc((size_t)B_ * H_ * 4);
    u16*   Wbf = (u16*)  alloc((size_t)3 * H_ * H_ * 2);
    float* xf  = (float*)alloc((size_t)BS_ * H_ * 4);
    u16*   xb  = (u16*)  alloc((size_t)BS_ * H_ * 2);
    u16*   qb  = (u16*)  alloc((size_t)BS_ * H_ * 2);
    u16*   kb  = (u16*)  alloc((size_t)BS_ * H_ * 2);
    u16*   vT  = (u16*)  alloc((size_t)BS_ * H_ * 2);

    float* pred = (float*)d_out;                 // [B,S]
    float* hid  = pred + (size_t)B_ * S_;        // [B,S,H]

    cvt_w_kernel<<<(H_ * H_ + 255) / 256, 256, 0, stream>>>(Wq, Wk, Wv, Wbf);
    cond_kernel<<<B_, 256, 0, stream>>>(relation, lhs, sub_head, sub_tail,
                                        rel_emb, W_rf, b_rf, gamma, beta, W_wd, W_bd, wgt, bia);
    ln_kernel<<<BS_, 256, 0, stream>>>(lhs, wgt, bia, xf, xb);
    qkv_kernel<<<dim3(BS_ / 16, 3), 128, 0, stream>>>(xb, Wbf, bq, bk, bv, qb, kb, vT);

    size_t smem = (size_t)4 * 16 * S_ * sizeof(float);   // 256 KB (<= 320 KB WGP LDS)
    hipFuncSetAttribute((const void*)attn_kernel,
                        hipFuncAttributeMaxDynamicSharedMemorySize, (int)smem);
    attn_kernel<<<dim3(S_ / 64, B_), 128, smem, stream>>>(qb, kb, vT, xf, mask,
                                                          W_obj, b_obj, pred, hid);
}
